// TransformerSER_24404004176368
// MI455X (gfx1250) — matrix operand order repaired and checked by reading, not compiled, not run on hardware
//
#include <hip/hip_runtime.h>
#include <hip/hip_bf16.h>
#include <math.h>

// ---------------- model constants ----------------
#define B_    16
#define NM_   80
#define TIN_  4000
#define C_    512
#define H_    8
#define L_    4
#define DFF_  2048
#define NC_   8
#define TP_   2000
#define DH_   64
#define MROWS 32000          // B_ * TP_
#define KCONV 256            // 240 padded to 256

typedef _Float16 half_t;
typedef __attribute__((ext_vector_type(16))) _Float16 v16h;
typedef __attribute__((ext_vector_type(8)))  _Float16 v8h;
typedef __attribute__((ext_vector_type(8)))  float    v8f;

union Frag16 { v16h v; struct { v8h lo, hi; } p; };

static __device__ __forceinline__ float gelu_exact(float x) {
    return 0.5f * x * (1.0f + erff(x * 0.70710678118654752440f));
}
static __device__ __forceinline__ float elu1(float x) {   // elu(x)+1
    return x > 0.0f ? x + 1.0f : __expf(x);
}

// CDNA5 async global->LDS copy (ASYNCcnt-tracked, no VGPR data path).
// Low 32 bits of a flat shared-aperture address == LDS byte offset (ISA 10.2).
static __device__ __forceinline__ void async_ld_b128(void* lds_p, const void* gptr) {
    unsigned lds = (unsigned)(uintptr_t)lds_p;
    asm volatile("global_load_async_to_lds_b128 %0, %1, off"
                 :: "v"(lds), "v"(gptr) : "memory");
}
static __device__ __forceinline__ void wait_async0() {
    asm volatile("s_wait_asynccnt 0" ::: "memory");
}

// =================================================================
// misc elementwise kernels
// =================================================================
__global__ void __launch_bounds__(256) cvt_f32_f16(const float* __restrict__ s,
                                                   half_t* __restrict__ d, int n) {
    int i = blockIdx.x * 256 + threadIdx.x;
    if (i < n) d[i] = (half_t)s[i];
}

__global__ void __launch_bounds__(256) pack_conv_w(const float* __restrict__ cw,
                                                   half_t* __restrict__ d) {
    int i = blockIdx.x * 256 + threadIdx.x;          // C_*KCONV elements
    int c = i >> 8, kk = i & 255;
    float v = (kk < NM_ * 3) ? cw[c * (NM_ * 3) + kk] : 0.0f;
    d[i] = (half_t)v;
}

// im2col for conv1d k=3, stride=2, pad=1: A[(b,t)][nm*3+j] = x[b][nm][2t-1+j]
__global__ void __launch_bounds__(256) im2col_k(const float* __restrict__ x,
                                                half_t* __restrict__ a) {
    long i = (long)blockIdx.x * 256 + threadIdx.x;   // MROWS*KCONV
    int kk = (int)(i & 255);
    int row = (int)(i >> 8);
    int b = row / TP_, t = row - b * TP_;
    float v = 0.0f;
    if (kk < NM_ * 3) {
        int nm = kk / 3, j = kk - nm * 3;
        int tin = 2 * t - 1 + j;
        if (tin >= 0 && tin < TIN_) v = x[((size_t)b * NM_ + nm) * TIN_ + tin];
    }
    a[i] = (half_t)v;
}

// =================================================================
// Generic 128x128 WMMA GEMM, BK=64, double-buffered async LDS staging.
//   MODE 1 CONV : gelu(acc*bn_scale + bn_b) + PE(t,n)      -> f32 h
//   MODE 2 QKV  : acc+bias, elu feature maps + mask        -> f16 q/k/v [B,H,Tp,DH]
//   MODE 3 RESID: acc+bias + outF32[row][n]                -> f32 (in-place residual)
//   MODE 4 FF1  : gelu(acc+bias)                           -> f16
// =================================================================
#define BM 128
#define BN 128
#define BK2 64
#define LDR2 72         // BK2 + 8 halfs padding

template<int MODE>
__global__ void __launch_bounds__(256) gemm_wmma(
    const half_t* __restrict__ A, const half_t* __restrict__ W,
    const float* __restrict__ bias,
    float* __restrict__ outF32, half_t* __restrict__ outF16,
    int K, int N,
    const float* __restrict__ aux0, const float* __restrict__ aux1,
    const int* __restrict__ lens,
    half_t* __restrict__ qf, half_t* __restrict__ kf, half_t* __restrict__ vv)
{
    __shared__ alignas(16) half_t As[2][BM * LDR2];
    __shared__ alignas(16) half_t Bs[2][BN * LDR2];

    const int tid  = threadIdx.x;
    const int wave = tid >> 5;
    const int lane = tid & 31;
    const int l15  = lane & 15;
    const int lh   = lane >> 4;
    const int wr   = wave >> 2;              // 0..1 : 64 rows
    const int wc   = wave & 3;               // 0..3 : 32 cols
    const int rowT = blockIdx.y * BM;
    const int colT = blockIdx.x * BN;

    const int ldr = tid >> 1;                // 0..127
    const int ldc = (tid & 1) * 32;          // 0 / 32

    v8f acc[4][2];
    for (int i = 0; i < 4; ++i)
        for (int j = 0; j < 2; ++j)
            for (int e = 0; e < 8; ++e) acc[i][j][e] = 0.0f;

    const half_t* ga = A + (size_t)(rowT + ldr) * K + ldc;
    const half_t* gw = W + (size_t)(colT + ldr) * K + ldc;

    // issue async tile copy (each thread: 32 halfs of A + 32 halfs of B)
    auto issue = [&](int buf, int k0) {
        half_t* la = &As[buf][ldr * LDR2 + ldc];
        half_t* lb = &Bs[buf][ldr * LDR2 + ldc];
        const half_t* pa = ga + k0;
        const half_t* pb = gw + k0;
        for (int u = 0; u < 4; ++u) {
            async_ld_b128(la + u * 8, pa + u * 8);
            async_ld_b128(lb + u * 8, pb + u * 8);
        }
    };

    const int nk = K / BK2;
    issue(0, 0);
    wait_async0();
    __syncthreads();

    for (int kt = 0; kt < nk; ++kt) {
        const int buf = kt & 1;
        if (kt + 1 < nk) issue(buf ^ 1, (kt + 1) * BK2);

        for (int ks = 0; ks < 2; ++ks) {          // two 16x16x32 k-steps
            Frag16 a[4], b[2];
            for (int i = 0; i < 4; ++i) {
                int r = wr * 64 + i * 16 + l15;
                int base = r * LDR2 + ks * 32;
                a[i].p.lo = *(const v8h*)&As[buf][base + lh * 8];
                a[i].p.hi = *(const v8h*)&As[buf][base + lh * 8 + 16];
            }
            for (int j = 0; j < 2; ++j) {
                int n = wc * 32 + j * 16 + l15;
                int base = n * LDR2 + ks * 32;
                b[j].p.lo = *(const v8h*)&Bs[buf][base + lh * 8];
                b[j].p.hi = *(const v8h*)&Bs[buf][base + lh * 8 + 16];
            }
            for (int i = 0; i < 4; ++i)
                for (int j = 0; j < 2; ++j)
                    acc[i][j] = __builtin_amdgcn_wmma_f32_16x16x32_f16(
                        false, a[i].v, false, b[j].v, (short)0, acc[i][j], false, false);
        }
        wait_async0();          // my async writes for tile kt+1 landed in LDS
        __syncthreads();        // everyone done reading buf / writing buf^1
    }

    // ---------------- epilogue ----------------
    for (int i = 0; i < 4; ++i) {
        for (int j = 0; j < 2; ++j) {
            const int nCol = colT + wc * 32 + j * 16 + l15;
            float bv = 0.0f;
            if (MODE != 1) bv = bias[nCol];
            for (int e = 0; e < 8; ++e) {
                const int row = rowT + wr * 64 + i * 16 + lh * 8 + e;
                float v = acc[i][j][e];
                if (MODE == 1) {            // conv + BN + GELU + PE
                    float sc = aux0[nCol] * rsqrtf(1.0f + 1e-5f);
                    float g  = gelu_exact(v * sc + aux1[nCol]);
                    int   t  = row % TP_;
                    int   i2 = nCol >> 1;
                    float fr = __expf(-(logf(10000.0f) / (float)C_) * (float)(2 * i2));
                    float an = (float)t * fr;
                    float pe = (nCol & 1) ? __cosf(an) : __sinf(an);
                    outF32[(size_t)row * N + nCol] = g + pe;
                } else if (MODE == 2) {     // QKV split + feature maps + mask
                    float val = v + bv;
                    int b2 = row / TP_, t = row - b2 * TP_;
                    int len2 = (lens[b2] + 1) >> 1;
                    float m = (t < len2) ? 1.0f : 0.0f;
                    int sec = nCol >> 9;            // /C_
                    int c   = nCol & (C_ - 1);
                    int hd  = c >> 6;               // /DH_
                    int d   = c & (DH_ - 1);
                    size_t dst = (((size_t)b2 * H_ + hd) * TP_ + t) * DH_ + d;
                    if      (sec == 0) qf[dst] = (half_t)elu1(val);
                    else if (sec == 1) kf[dst] = (half_t)(m * elu1(val));
                    else               vv[dst] = (half_t)(val * m);
                } else if (MODE == 3) {     // bias + residual, in-place f32
                    size_t idx = (size_t)row * N + nCol;
                    outF32[idx] = outF32[idx] + v + bv;
                } else {                    // FF1: gelu -> f16
                    outF16[(size_t)row * N + nCol] = (half_t)gelu_exact(v + bv);
                }
            }
        }
    }
}

// =================================================================
// linear attention:  kv[d][e] = sum_n kf[n][d]*v[n][e],  z[d] = sum_n kf[n][d]
// one block per (b,h); M=N=64, K=Tp, WMMA 16x16x32 with transposed LDS staging
// =================================================================
#define LDA 40
__global__ void __launch_bounds__(256) attn_kv_kernel(
    const half_t* __restrict__ kf, const half_t* __restrict__ vv,
    float* __restrict__ kv, float* __restrict__ z)
{
    const int bh = blockIdx.x;
    const half_t* kfb = kf + (size_t)bh * TP_ * DH_;
    const half_t* vb  = vv + (size_t)bh * TP_ * DH_;
    __shared__ alignas(16) half_t Kt[DH_ * LDA];   // [d][n]
    __shared__ alignas(16) half_t Vt[DH_ * LDA];   // [e][n]
    __shared__ float zred[256];

    const int tid  = threadIdx.x;
    const int wave = tid >> 5;
    const int lane = tid & 31;
    const int l15  = lane & 15, lh = lane >> 4;
    const int tr   = wave >> 1;               // d-tile 0..3
    const int tc0  = (wave & 1) * 2;          // e-tiles {0,1} or {2,3}

    const int ldn = tid >> 3;                 // n within 32-chunk
    const int ldd = (tid & 7) * 8;            // d group

    v8f acc[2];
    for (int j = 0; j < 2; ++j) for (int e = 0; e < 8; ++e) acc[j][e] = 0.0f;
    float zpart = 0.0f;

    for (int n0 = 0; n0 < TP_; n0 += 32) {
        v8h kvk; v8h kvv2;
        for (int u = 0; u < 8; ++u) { kvk[u] = (_Float16)0; kvv2[u] = (_Float16)0; }
        if (n0 + ldn < TP_) {
            kvk  = *(const v8h*)&kfb[(size_t)(n0 + ldn) * DH_ + ldd];
            kvv2 = *(const v8h*)&vb [(size_t)(n0 + ldn) * DH_ + ldd];
        }
        for (int u = 0; u < 8; ++u) {
            Kt[(ldd + u) * LDA + ldn] = kvk[u];
            Vt[(ldd + u) * LDA + ldn] = kvv2[u];
        }
        __syncthreads();

        { // z partial: d = tid&63, quarter = tid>>6 over 8 n's
            int d = tid & 63, q = tid >> 6;
            for (int u = 0; u < 8; ++u) zpart += (float)Kt[d * LDA + q * 8 + u];
        }
        Frag16 a;
        a.p.lo = *(const v8h*)&Kt[(tr * 16 + l15) * LDA + lh * 8];
        a.p.hi = *(const v8h*)&Kt[(tr * 16 + l15) * LDA + lh * 8 + 16];
        for (int j = 0; j < 2; ++j) {
            Frag16 b;
            int e = (tc0 + j) * 16 + l15;
            b.p.lo = *(const v8h*)&Vt[e * LDA + lh * 8];
            b.p.hi = *(const v8h*)&Vt[e * LDA + lh * 8 + 16];
            acc[j] = __builtin_amdgcn_wmma_f32_16x16x32_f16(
                false, a.v, false, b.v, (short)0, acc[j], false, false);
        }
        __syncthreads();
    }
    for (int j = 0; j < 2; ++j)
        for (int u = 0; u < 8; ++u) {
            int d = tr * 16 + lh * 8 + u;
            int e = (tc0 + j) * 16 + l15;
            kv[((size_t)bh * DH_ + d) * DH_ + e] = acc[j][u];
        }
    zred[tid] = zpart;
    __syncthreads();
    if (tid < 64)
        z[bh * DH_ + tid] = zred[tid] + zred[tid + 64] + zred[tid + 128] + zred[tid + 192];
}

// =================================================================
// y[t][e] = (qf[t][:] @ kv[:,e]) / max(qf[t][:]·z, eps) -> ybuf[B,Tp,C] f16
// grid (ceil(Tp/128), B*H); M-tile 128, N=64, K=64
// =================================================================
#define LDY 72
__global__ void __launch_bounds__(256) attn_y_kernel(
    const half_t* __restrict__ qf, const float* __restrict__ kv,
    const float* __restrict__ z, half_t* __restrict__ ybuf)
{
    const int bh = blockIdx.y;
    const int b  = bh >> 3, hd = bh & 7;
    const int t0 = blockIdx.x * 128;
    __shared__ alignas(16) half_t Aq[128 * LDY];
    __shared__ alignas(16) half_t Bk[DH_ * LDY];   // [e][d]
    __shared__ float zs[DH_];
    __shared__ float den[128];

    const int tid = threadIdx.x;
    const int wave = tid >> 5;
    const int lane = tid & 31;
    const int l15 = lane & 15, lh = lane >> 4;

    const half_t* qb = qf + (size_t)bh * TP_ * DH_;

    { // stage qf rows (zero-pad past Tp)
        int r = tid >> 1, c0 = (tid & 1) * 32;
        for (int u = 0; u < 4; ++u) {
            v8h v;
            for (int w = 0; w < 8; ++w) v[w] = (_Float16)0;
            if (t0 + r < TP_)
                v = *(const v8h*)&qb[(size_t)(t0 + r) * DH_ + c0 + u * 8];
            *(v8h*)&Aq[r * LDY + c0 + u * 8] = v;
        }
    }
    { // stage kv transposed -> Bk[e][d]
        for (int i = 0; i < 16; ++i) {
            int lin = tid + 256 * i;               // 4096 elements
            int d = lin >> 6, e = lin & 63;
            Bk[e * LDY + d] = (half_t)kv[((size_t)bh * DH_ + d) * DH_ + e];
        }
        if (tid < DH_) zs[tid] = z[bh * DH_ + tid];
    }
    __syncthreads();

    if (tid < 128) {
        float s = 0.0f;
        for (int d = 0; d < DH_; ++d) s += (float)Aq[tid * LDY + d] * zs[d];
        den[tid] = fmaxf(s, 1e-6f);
    }
    __syncthreads();

    v8f acc[4];
    for (int j = 0; j < 4; ++j) for (int e = 0; e < 8; ++e) acc[j][e] = 0.0f;
    for (int ks = 0; ks < 2; ++ks) {
        Frag16 a;
        int r = wave * 16 + l15;
        a.p.lo = *(const v8h*)&Aq[r * LDY + ks * 32 + lh * 8];
        a.p.hi = *(const v8h*)&Aq[r * LDY + ks * 32 + lh * 8 + 16];
        for (int j = 0; j < 4; ++j) {
            Frag16 bfr;
            int e = j * 16 + l15;
            bfr.p.lo = *(const v8h*)&Bk[e * LDY + ks * 32 + lh * 8];
            bfr.p.hi = *(const v8h*)&Bk[e * LDY + ks * 32 + lh * 8 + 16];
            acc[j] = __builtin_amdgcn_wmma_f32_16x16x32_f16(
                false, a.v, false, bfr.v, (short)0, acc[j], false, false);
        }
    }
    for (int j = 0; j < 4; ++j)
        for (int u = 0; u < 8; ++u) {
            int r = wave * 16 + lh * 8 + u;
            int t = t0 + r;
            int e = j * 16 + l15;
            if (t < TP_)
                ybuf[((size_t)b * TP_ + t) * C_ + hd * DH_ + e] =
                    (half_t)(acc[j][u] / den[r]);
        }
}

// =================================================================
// LayerNorm over C=512: one block per row, writes f16
// =================================================================
__global__ void __launch_bounds__(256) ln_kernel(
    const float* __restrict__ hbuf, const float* __restrict__ g,
    const float* __restrict__ beta, half_t* __restrict__ out)
{
    const int row = blockIdx.x;
    const int tid = threadIdx.x;
    const float* xr = hbuf + (size_t)row * C_;
    float v0 = xr[tid], v1 = xr[tid + 256];
    __shared__ float red[8];

    float s = v0 + v1;
    for (int o = 16; o; o >>= 1) s += __shfl_down(s, o, 32);
    if ((tid & 31) == 0) red[tid >> 5] = s;
    __syncthreads();
    float mean = 0.0f;
    for (int i = 0; i < 8; ++i) mean += red[i];
    mean *= (1.0f / C_);
    __syncthreads();

    float d0 = v0 - mean, d1 = v1 - mean;
    float q = d0 * d0 + d1 * d1;
    for (int o = 16; o; o >>= 1) q += __shfl_down(q, o, 32);
    if ((tid & 31) == 0) red[tid >> 5] = q;
    __syncthreads();
    float var = 0.0f;
    for (int i = 0; i < 8; ++i) var += red[i];
    var *= (1.0f / C_);
    float rs = rsqrtf(var + 1e-5f);

    half_t* o2 = out + (size_t)row * C_;
    o2[tid]       = (half_t)(d0 * rs * g[tid] + beta[tid]);
    o2[tid + 256] = (half_t)(d1 * rs * g[tid + 256] + beta[tid + 256]);
}

// =================================================================
// masked mean pool (two phase) + FC
// =================================================================
#define PSL 8
__global__ void __launch_bounds__(256) pool_partial(
    const float* __restrict__ h, const int* __restrict__ lens,
    float* __restrict__ part)
{
    const int s = blockIdx.x, b = blockIdx.y;
    const int c = threadIdx.x;
    int len2 = (lens[b] + 1) >> 1;
    if (len2 > TP_) len2 = TP_;
    float a0 = 0.0f, a1 = 0.0f;
    const int t0 = s * (TP_ / PSL), t1 = t0 + (TP_ / PSL);
    for (int t = t0; t < t1; ++t) {
        if (t < len2) {
            const float* row = h + ((size_t)b * TP_ + t) * C_;
            a0 += row[c];
            a1 += row[c + 256];
        }
    }
    float* p = part + ((size_t)b * PSL + s) * C_;
    p[c] = a0;
    p[c + 256] = a1;
}

__global__ void __launch_bounds__(256) pool_fc(
    const float* __restrict__ part, const int* __restrict__ lens,
    const float* __restrict__ fw, const float* __restrict__ fb,
    float* __restrict__ out)
{
    const int b = blockIdx.x;
    const int tid = threadIdx.x;
    __shared__ float pooled[C_];
    int len2 = (lens[b] + 1) >> 1;
    if (len2 > TP_) len2 = TP_;
    float inv = 1.0f / fmaxf((float)len2, 1.0f);
    for (int c = tid; c < C_; c += 256) {
        float s = 0.0f;
        for (int p = 0; p < PSL; ++p) s += part[((size_t)b * PSL + p) * C_ + c];
        pooled[c] = s * inv;
    }
    __syncthreads();
    int w = tid >> 5, lane = tid & 31;   // wave w -> class w (NC_==8 waves)
    float s = 0.0f;
    for (int i = 0; i < C_ / 32; ++i) {
        int c = lane + i * 32;
        s += pooled[c] * fw[w * C_ + c];
    }
    for (int o = 16; o; o >>= 1) s += __shfl_down(s, o, 32);
    if (lane == 0) out[b * NC_ + w] = s + fb[w];
}

// =================================================================
// host launcher
// =================================================================
extern "C" void kernel_launch(void* const* d_in, const int* in_sizes, int n_in,
                              void* d_out, int out_size, void* d_ws, size_t ws_size,
                              hipStream_t stream) {
    const float* x      = (const float*)d_in[0];
    const int*   lens   = (const int*)  d_in[1];
    const float* conv_w = (const float*)d_in[2];
    const float* bn_g   = (const float*)d_in[3];
    const float* bn_b   = (const float*)d_in[4];
    const float* qkv_w  = (const float*)d_in[5];
    const float* qkv_b  = (const float*)d_in[6];
    const float* out_w  = (const float*)d_in[7];
    const float* out_b  = (const float*)d_in[8];
    const float* lin1_w = (const float*)d_in[9];
    const float* lin1_b = (const float*)d_in[10];
    const float* lin2_w = (const float*)d_in[11];
    const float* lin2_b = (const float*)d_in[12];
    const float* ln1_g  = (const float*)d_in[13];
    const float* ln1_b  = (const float*)d_in[14];
    const float* ln2_g  = (const float*)d_in[15];
    const float* ln2_b  = (const float*)d_in[16];
    const float* fc_w   = (const float*)d_in[17];
    const float* fc_b   = (const float*)d_in[18];
    float* out = (float*)d_out;

    // ---- workspace layout (bytes) ----
    char* ws = (char*)d_ws;
    constexpr size_t SZ_H     = (size_t)MROWS * C_ * 4;
    constexpr size_t SZ_HN    = (size_t)MROWS * C_ * 2;
    constexpr size_t SZ_IM    = (size_t)MROWS * KCONV * 2;
    constexpr size_t SZ_WCONV = (size_t)C_ * KCONV * 2;
    constexpr size_t SZ_WQKV  = (size_t)L_ * 3 * C_ * C_ * 2;
    constexpr size_t SZ_WOUT  = (size_t)L_ * C_ * C_ * 2;
    constexpr size_t SZ_WL1   = (size_t)L_ * DFF_ * C_ * 2;
    constexpr size_t SZ_WL2   = (size_t)L_ * C_ * DFF_ * 2;
    constexpr size_t SZ_KV    = (size_t)B_ * H_ * DH_ * DH_ * 4;
    constexpr size_t SZ_Z     = (size_t)B_ * H_ * DH_ * 4;
    constexpr size_t SZ_PART  = (size_t)B_ * PSL * C_ * 4;

    size_t off = 0;
    auto take = [&](size_t sz) { size_t o = off; off += (sz + 255) & ~(size_t)255; return o; };
    const size_t OFF_H     = take(SZ_H);
    const size_t OFF_HN    = take(SZ_HN);
    const size_t OFF_IM    = take(SZ_IM);
    const size_t OFF_WCONV = take(SZ_WCONV);
    const size_t OFF_WQKV  = take(SZ_WQKV);
    const size_t OFF_WOUT  = take(SZ_WOUT);
    const size_t OFF_WL1   = take(SZ_WL1);
    const size_t OFF_WL2   = take(SZ_WL2);
    const size_t OFF_KV    = take(SZ_KV);
    const size_t OFF_Z     = take(SZ_Z);
    const size_t OFF_PART  = take(SZ_PART);
    const size_t OFF_ATT   = take(0);   // attention region, overlapped with FFN scratch

    float*  h    = (float*) (ws + OFF_H);
    half_t* hn   = (half_t*)(ws + OFF_HN);
    half_t* im   = (half_t*)(ws + OFF_IM);
    half_t* wconv= (half_t*)(ws + OFF_WCONV);
    half_t* wqkv = (half_t*)(ws + OFF_WQKV);
    half_t* wout = (half_t*)(ws + OFF_WOUT);
    half_t* wl1  = (half_t*)(ws + OFF_WL1);
    half_t* wl2  = (half_t*)(ws + OFF_WL2);
    float*  kvb  = (float*) (ws + OFF_KV);
    float*  zb   = (float*) (ws + OFF_Z);
    float*  part = (float*) (ws + OFF_PART);
    half_t* qfb  = (half_t*)(ws + OFF_ATT);
    half_t* kfb  = qfb + (size_t)MROWS * C_;
    half_t* vvb  = kfb + (size_t)MROWS * C_;
    half_t* ybuf = vvb + (size_t)MROWS * C_;
    half_t* ffb  = qfb;                 // FFN intermediate reuses attention region
    (void)ws_size; (void)in_sizes; (void)n_in; (void)out_size;

    // ---- weight conversion ----
    auto cvt = [&](const float* s, half_t* d, int n) {
        cvt_f32_f16<<<dim3((n + 255) / 256), 256, 0, stream>>>(s, d, n);
    };
    cvt(qkv_w,  wqkv, L_ * 3 * C_ * C_);
    cvt(out_w,  wout, L_ * C_ * C_);
    cvt(lin1_w, wl1,  L_ * DFF_ * C_);
    cvt(lin2_w, wl2,  L_ * C_ * DFF_);
    pack_conv_w<<<dim3((C_ * KCONV) / 256), 256, 0, stream>>>(conv_w, wconv);
    im2col_k<<<dim3((MROWS * KCONV) / 256), 256, 0, stream>>>(x, im);

    // ---- conv-as-GEMM + BN + GELU + PE -> h ----
    gemm_wmma<1><<<dim3(C_ / BN, MROWS / BM), 256, 0, stream>>>(
        im, wconv, nullptr, h, nullptr, KCONV, C_,
        bn_g, bn_b, nullptr, nullptr, nullptr, nullptr);

    // ---- transformer layers ----
    for (int l = 0; l < L_; ++l) {
        const half_t* Wq = wqkv + (size_t)l * 3 * C_ * C_;
        const half_t* Wo = wout + (size_t)l * C_ * C_;
        const half_t* W1 = wl1  + (size_t)l * DFF_ * C_;
        const half_t* W2 = wl2  + (size_t)l * C_ * DFF_;

        ln_kernel<<<dim3(MROWS), 256, 0, stream>>>(h, ln1_g + l * C_, ln1_b + l * C_, hn);

        gemm_wmma<2><<<dim3(3 * C_ / BN, MROWS / BM), 256, 0, stream>>>(
            hn, Wq, qkv_b + (size_t)l * 3 * C_, nullptr, nullptr, C_, 3 * C_,
            nullptr, nullptr, lens, qfb, kfb, vvb);

        attn_kv_kernel<<<dim3(B_ * H_), 256, 0, stream>>>(kfb, vvb, kvb, zb);
        attn_y_kernel<<<dim3((TP_ + 127) / 128, B_ * H_), 256, 0, stream>>>(qfb, kvb, zb, ybuf);

        gemm_wmma<3><<<dim3(C_ / BN, MROWS / BM), 256, 0, stream>>>(
            ybuf, Wo, out_b + (size_t)l * C_, h, nullptr, C_, C_,
            nullptr, nullptr, nullptr, nullptr, nullptr, nullptr);

        ln_kernel<<<dim3(MROWS), 256, 0, stream>>>(h, ln2_g + l * C_, ln2_b + l * C_, hn);

        gemm_wmma<4><<<dim3(DFF_ / BN, MROWS / BM), 256, 0, stream>>>(
            hn, W1, lin1_b + (size_t)l * DFF_, nullptr, ffb, C_, DFF_,
            nullptr, nullptr, nullptr, nullptr, nullptr, nullptr);

        gemm_wmma<3><<<dim3(C_ / BN, MROWS / BM), 256, 0, stream>>>(
            ffb, W2, lin2_b + (size_t)l * C_, h, nullptr, DFF_, C_,
            nullptr, nullptr, nullptr, nullptr, nullptr, nullptr);
    }

    // ---- pool + FC ----
    pool_partial<<<dim3(PSL, B_), 256, 0, stream>>>(h, lens, part);
    pool_fc<<<dim3(B_), 256, 0, stream>>>(part, lens, fc_w, fc_b, out);
}
